// ChannelReasonModule_74388833567126
// MI455X (gfx1250) — matrix operand order repaired and checked by reading, not compiled, not run on hardware
//
#include <hip/hip_runtime.h>
#include <hip/hip_bf16.h>
#include <stddef.h>
#include <stdint.h>

// ---------------------------------------------------------------------------
// ChannelReasonModule fused pipeline for MI455X (gfx1250, wave32, WMMA).
//
// b=4, cin=512, planes=256, HW=16384. GROUPS==planes -> att is a per-channel
// spatial dot product; GroupNorm stats are exact functions of per-channel
// sum(t), sum(t^2) because every output channel is a scalar multiple of t.
//
// k_wpack: pre-split Wt|Wp|Wg into blocked bf16 hi/lo planes (LDS-tile layout).
// k_gemm : fused t/p/g GEMM, bf16 hi/lo split (3 WMMAs / K=32 chunk ~ f32
//          accuracy), 36 v_wmma per wave per K-step; att/S1/S2 reduced
//          in-register (shuffles + f32 atomics); t staged into d_out.
//          A tiles staged via async global->LDS (ASYNCcnt) when available.
// k_graph: tiny graph network + analytic GroupNorm constants.
// k_final: out[b,2c+j,:] = s_j * t + o_j + x   (float4).
// ---------------------------------------------------------------------------

typedef __attribute__((ext_vector_type(16))) __bf16 v16bf;
typedef __attribute__((ext_vector_type(8)))  __bf16 v8bf;
typedef __attribute__((ext_vector_type(2)))  __bf16 bf16x2;
typedef __attribute__((ext_vector_type(8)))  float  v8f;
typedef __attribute__((ext_vector_type(2)))  int    v2i;

#define HW      16384
#define CIN     512
#define PLANES  256
#define NTILE   256   // spatial columns per block
#define CTILE   32    // channels per block (per each of t/p/g)
#define KTILE   32    // K chunk (one bf16 WMMA depth)
#define LDSS    40    // padded LDS row stride in bf16 elements (80 B)
#define WCHUNK  (96 * KTILE)  // one (cblk,kc) weight chunk: 96 rows x 32 k

#if defined(__has_builtin)
#if __has_builtin(__builtin_amdgcn_global_load_async_to_lds_b64)
#define HAVE_ASYNC_LDS 1
#endif
#endif
#ifndef HAVE_ASYNC_LDS
#define HAVE_ASYNC_LDS 0
#endif

typedef __attribute__((address_space(1))) v2i as1_v2i;
typedef __attribute__((address_space(3))) v2i as3_v2i;

__device__ __forceinline__ void async_copy_b64(const __bf16* g, __bf16* l) {
#if HAVE_ASYNC_LDS
  __builtin_amdgcn_global_load_async_to_lds_b64(
      (as1_v2i*)(void*)g, (as3_v2i*)(void*)l, 0, 0);
#else
  *(uint64_t*)l = *(const uint64_t*)g;
#endif
}

__device__ __forceinline__ void wait_async_lds() {
#if HAVE_ASYNC_LDS
  asm volatile("s_wait_asynccnt 0x0" ::: "memory");
#endif
}

__device__ __forceinline__ v8f vzero8() {
  v8f z;
#pragma unroll
  for (int i = 0; i < 8; ++i) z[i] = 0.0f;
  return z;
}

__device__ __forceinline__ v16bf mk16(v8bf lo, v8bf hi) {
  v16bf r;
#pragma unroll
  for (int i = 0; i < 8; ++i) { r[i] = lo[i]; r[i + 8] = hi[i]; }
  return r;
}

__device__ __forceinline__ v8f wmma_bf16(v16bf a, v16bf b, v8f c) {
  return __builtin_amdgcn_wmma_f32_16x16x32_bf16(false, a, false, b,
                                                 (short)0, c, false, false);
}

// ---------------------------------------------------------------------------
// Kernel 0: zero att/S1/S2 accumulators (3 * 1024 floats, contiguous).
// ---------------------------------------------------------------------------
__global__ void k_zero(float* __restrict__ p, int n) {
  int i = blockIdx.x * 256 + threadIdx.x;
  if (i < n) p[i] = 0.0f;
}

// ---------------------------------------------------------------------------
// Kernel W: split weights into blocked bf16 hi/lo planes.
// Plane layout: [(cblk*16 + kc)][row 0..95][k 0..31], linear; rows are
// mat*32 + c with mat in {t,p,g}. One thread = one K pair; linear dst = q*2.
// ---------------------------------------------------------------------------
__global__ __launch_bounds__(256) void k_wpack(
    const float* __restrict__ Wt, const float* __restrict__ Wp,
    const float* __restrict__ Wg, __bf16* __restrict__ Whi,
    __bf16* __restrict__ Wlo) {
  const int q     = blockIdx.x * 256 + threadIdx.x;  // 0..196607
  const int kp    = q & 15;
  const int row   = (q >> 4) % 96;
  const int chunk = q / 1536;        // cblk*16 + kc
  const int kc    = chunk & 15;
  const int cblk  = chunk >> 4;
  const int mat   = row >> 5;
  const int c     = row & 31;
  const float* W = (mat == 0) ? Wt : ((mat == 1) ? Wp : Wg);
  const float2 f = *(const float2*)&W[(size_t)(cblk * 32 + c) * CIN + kc * 32 + kp * 2];
  const __bf16 h0 = (__bf16)f.x, h1 = (__bf16)f.y;
  bf16x2 hi; hi[0] = h0; hi[1] = h1;
  bf16x2 lo; lo[0] = (__bf16)(f.x - (float)h0);
  lo[1] = (__bf16)(f.y - (float)h1);
  *(bf16x2*)&Whi[q * 2] = hi;
  *(bf16x2*)&Wlo[q * 2] = lo;
}

// ---------------------------------------------------------------------------
// Kernel 1: fused triple GEMM + reductions.
// grid = (HW/NTILE, PLANES/CTILE, b) ; block = 256 threads = 8 waves.
// Wave w owns 2 column fragments {2w, 2w+1}; 12 accumulator tiles:
// {t,p,g} x {mfrag 0,1} x {nfrag 0,1}; 36 WMMAs per K-step.
// ---------------------------------------------------------------------------
__global__ __launch_bounds__(256) void k_gemm(
    const float* __restrict__ x,
    const __bf16* __restrict__ Whi, const __bf16* __restrict__ Wlo,
    float* __restrict__ t_out,            // == d_out, even-channel slots
    float* __restrict__ att, float* __restrict__ S1, float* __restrict__ S2) {
  __shared__ __bf16 Bhi[NTILE * LDSS];          // 20 KB
  __shared__ __bf16 Blo[NTILE * LDSS];          // 20 KB
  __shared__ __bf16 Ahi[3 * CTILE * LDSS];      // 7.5 KB
  __shared__ __bf16 Alo[3 * CTILE * LDSS];      // 7.5 KB

  const int tid  = threadIdx.x;
  const int n0   = blockIdx.x * NTILE;
  const int c0   = blockIdx.y * CTILE;
  const int b    = blockIdx.z;
  const int wave = tid >> 5;
  const int lane = tid & 31;
  const float* xb = x + (size_t)b * CIN * HW;

  v8f acc[3][2][2];
#pragma unroll
  for (int m = 0; m < 3; ++m)
#pragma unroll
    for (int f = 0; f < 2; ++f)
#pragma unroll
      for (int nf = 0; nf < 2; ++nf) acc[m][f][nf] = vzero8();

  for (int k0 = 0; k0 < CIN; k0 += KTILE) {
    // ---- A: async-copy pre-packed weight chunk into padded LDS rows -------
    {
      const __bf16* gih = Whi + (size_t)(blockIdx.y * 16 + (k0 >> 5)) * WCHUNK;
      const __bf16* gil = Wlo + (size_t)(blockIdx.y * 16 + (k0 >> 5)) * WCHUNK;
#pragma unroll
      for (int i = 0; i < 3; ++i) {
        const int qq  = i * 256 + tid;  // quad of 4 bf16, 0..767
        const int row = qq >> 3;
        const int kq  = qq & 7;
        async_copy_b64(gih + qq * 4, &Ahi[row * LDSS + kq * 4]);
        async_copy_b64(gil + qq * 4, &Alo[row * LDSS + kq * 4]);
      }
    }
    // ---- B: x[k0:k0+32, n0:n0+256] -> hi/lo bf16, layout [n][k] ----------
    {
      const int n4 = (tid & 63) * 4;
      const int kg = tid >> 6;  // 0..3
#pragma unroll
      for (int i = 0; i < 4; ++i) {
        const int kp = kg * 4 + i;
        const int k  = kp * 2;
        const float4 f0 = *(const float4*)&xb[(size_t)(k0 + k) * HW + n0 + n4];
        const float4 f1 = *(const float4*)&xb[(size_t)(k0 + k + 1) * HW + n0 + n4];
        const float a0[4] = {f0.x, f0.y, f0.z, f0.w};
        const float a1[4] = {f1.x, f1.y, f1.z, f1.w};
#pragma unroll
        for (int j = 0; j < 4; ++j) {
          const __bf16 h0 = (__bf16)a0[j], h1 = (__bf16)a1[j];
          bf16x2 hi; hi[0] = h0; hi[1] = h1;
          bf16x2 lo; lo[0] = (__bf16)(a0[j] - (float)h0);
          lo[1] = (__bf16)(a1[j] - (float)h1);
          *(bf16x2*)&Bhi[(n4 + j) * LDSS + k] = hi;
          *(bf16x2*)&Blo[(n4 + j) * LDSS + k] = lo;
        }
      }
    }
    wait_async_lds();
    __syncthreads();

    // ---- B fragments: lane<16 -> K 0..15 ; lane>=16 -> K 16..31 ----------
    const int ncol = lane & 15;
    const int boff = (lane < 16) ? 0 : 16;
    v16bf bhi[2], blo[2];
#pragma unroll
    for (int nf = 0; nf < 2; ++nf) {
      const __bf16* bp  = &Bhi[((wave * 2 + nf) * 16 + ncol) * LDSS + boff];
      const __bf16* bpl = &Blo[((wave * 2 + nf) * 16 + ncol) * LDSS + boff];
      bhi[nf] = mk16(*(const v8bf*)bp,  *(const v8bf*)(bp + 8));
      blo[nf] = mk16(*(const v8bf*)bpl, *(const v8bf*)(bpl + 8));
    }

    // ---- A fragments + 3 WMMAs (hi*hi + hi*lo + lo*hi) per tile ----------
    const int arow = lane & 15;
    const int aoff = (lane < 16) ? 0 : 16;  // K 0..15 vs 16..31
#pragma unroll
    for (int mat = 0; mat < 3; ++mat) {
#pragma unroll
      for (int mf = 0; mf < 2; ++mf) {
        const int row = mat * CTILE + mf * 16 + arow;
        const __bf16* ap  = &Ahi[row * LDSS + aoff];
        const __bf16* apl = &Alo[row * LDSS + aoff];
        const v16bf ahi = mk16(*(const v8bf*)ap,  *(const v8bf*)(ap + 8));
        const v16bf alo = mk16(*(const v8bf*)apl, *(const v8bf*)(apl + 8));
#pragma unroll
        for (int nf = 0; nf < 2; ++nf) {
          v8f c = acc[mat][mf][nf];
          c = wmma_bf16(ahi, bhi[nf], c);
          c = wmma_bf16(ahi, blo[nf], c);
          c = wmma_bf16(alo, bhi[nf], c);
          acc[mat][mf][nf] = c;
        }
      }
    }
    __syncthreads();
  }

  // ---- epilogue: store t into d_out even-channel slots; reduce att/S1/S2 --
  // C/D layout: VGPR r, lane<16 -> (M=r, N=lane) ; lane>=16 -> (M=r+8, N=lane-16)
  const int mbase = (lane < 16) ? 0 : 8;
#pragma unroll
  for (int nf = 0; nf < 2; ++nf) {
    const int col = n0 + (wave * 2 + nf) * 16 + (lane & 15);
#pragma unroll
    for (int mf = 0; mf < 2; ++mf) {
#pragma unroll
      for (int r = 0; r < 8; ++r) {
        const int c = c0 + mf * 16 + mbase + r;
        // t channel c lives at out channel 2c (read back by kernel 3)
        t_out[((size_t)b * 2 * PLANES + 2 * c) * HW + col] = acc[0][mf][nf][r];
        float s1 = acc[0][mf][nf][r];
        float s2 = s1 * s1;
        float pg = acc[1][mf][nf][r] * acc[2][mf][nf][r];
#pragma unroll
        for (int off = 1; off < 16; off <<= 1) {
          s1 += __shfl_xor(s1, off);
          s2 += __shfl_xor(s2, off);
          pg += __shfl_xor(pg, off);
        }
        if ((lane & 15) == 0) {
          atomicAdd(&S1[b * PLANES + c], s1);
          atomicAdd(&S2[b * PLANES + c], s2);
          atomicAdd(&att[b * PLANES + c], pg);
        }
      }
    }
  }
}

// ---------------------------------------------------------------------------
// Kernel 2: graph network + analytic GroupNorm constants. One block.
// ---------------------------------------------------------------------------
__global__ __launch_bounds__(256) void k_graph(
    const float* __restrict__ att, const float* __restrict__ S1,
    const float* __restrict__ S2, const float* __restrict__ Wz,
    const float* __restrict__ W_adj, const float* __restrict__ g_adj,
    const float* __restrict__ b_adj, const float* __restrict__ m_adj,
    const float* __restrict__ v_adj, const float* __restrict__ W_wg,
    const float* __restrict__ g_wg, const float* __restrict__ b_wg,
    const float* __restrict__ m_wg, const float* __restrict__ v_wg,
    const float* __restrict__ gn_g, const float* __restrict__ gn_b,
    float* __restrict__ params) {
  __shared__ float zi[4 * 32 * 8];
  __shared__ float z1[4 * 32 * 8];
  const int tid = threadIdx.x;
  const float eps = 1e-5f;

  for (int e = tid; e < 1024; e += 256) {
    const int b = e >> 8, i = (e >> 3) & 31, k = e & 7;
    zi[e] = att[b * PLANES + k * 32 + i];   // z_idt[b,i,k] = att[b, k*32+i]
  }
  __syncthreads();
  for (int e = tid; e < 1024; e += 256) {
    const int b = e >> 8, i = (e >> 3) & 31, k = e & 7;
    float s = 0.0f;
#pragma unroll 8
    for (int m = 0; m < 32; ++m) s += W_adj[i * 32 + m] * zi[(b * 32 + m) * 8 + k];
    s = (s - m_adj[i]) * rsqrtf(v_adj[i] + eps) * g_adj[i] + b_adj[i];
    z1[e] = fmaxf(s, 0.0f) + zi[e];
  }
  __syncthreads();
  for (int e = tid; e < 1024; e += 256) {
    const int b = e >> 8, i = (e >> 3) & 31, k = e & 7;
    float s = 0.0f;
#pragma unroll
    for (int f = 0; f < 8; ++f) s += W_wg[k * 8 + f] * z1[(b * 32 + i) * 8 + f];
    s = (s - m_wg[k]) * rsqrtf(v_wg[k] + eps) * g_wg[k] + b_wg[k];
    const float a = fmaxf(s, 0.0f);          // zvec[b, c],  c = i*8 + k
    const int   c = i * 8 + k;
    const float w0 = Wz[2 * c], w1 = Wz[2 * c + 1];
    const float s1 = S1[b * PLANES + c], s2 = S2[b * PLANES + c];
    const float inv2hw = 1.0f / (2.0f * (float)HW);
    const float mean = a * (w0 + w1) * s1 * inv2hw;
    const float e2   = a * a * (w0 * w0 + w1 * w1) * s2 * inv2hw;
    const float rstd = rsqrtf(e2 - mean * mean + eps);
    const float g0 = gn_g[2 * c], g1 = gn_g[2 * c + 1];
    float* pc = &params[(b * PLANES + c) * 4];
    pc[0] = w0 * a * rstd * g0;
    pc[1] = gn_b[2 * c] - mean * rstd * g0;
    pc[2] = w1 * a * rstd * g1;
    pc[3] = gn_b[2 * c + 1] - mean * rstd * g1;
  }
}

// ---------------------------------------------------------------------------
// Kernel 3: out[b,2c+j,:] = s_j * t + o_j + x[b,2c+j,:]  (float4 vectorized).
// t lives in out[b,2c,:] (written by kernel 1); each thread reads its own
// slot before overwriting it, so the in-place staging is race-free.
// ---------------------------------------------------------------------------
__global__ __launch_bounds__(256) void k_final(
    const float* __restrict__ x, const float* __restrict__ params,
    float* __restrict__ out) {
  const size_t tid = (size_t)blockIdx.x * 256 + threadIdx.x;  // 2^22 threads
  const int q = (int)(tid & 4095);
  const int c = (int)((tid >> 12) & 255);
  const int b = (int)(tid >> 20);
  const size_t base0 = (((size_t)b * 512 + 2 * c) << 14) + ((size_t)q << 2);
  const size_t base1 = base0 + HW;
  const float4 tv = *(const float4*)&out[base0];  // staged t
  const float* pc = &params[((b << 8) + c) << 2];
  const float s0 = pc[0], o0 = pc[1], s1 = pc[2], o1 = pc[3];
  const float4 x0 = *(const float4*)&x[base0];
  const float4 x1 = *(const float4*)&x[base1];
  float4 r0, r1;
  r0.x = s0 * tv.x + o0 + x0.x;  r0.y = s0 * tv.y + o0 + x0.y;
  r0.z = s0 * tv.z + o0 + x0.z;  r0.w = s0 * tv.w + o0 + x0.w;
  r1.x = s1 * tv.x + o1 + x1.x;  r1.y = s1 * tv.y + o1 + x1.y;
  r1.z = s1 * tv.z + o1 + x1.z;  r1.w = s1 * tv.w + o1 + x1.w;
  *(float4*)&out[base0] = r0;
  *(float4*)&out[base1] = r1;
}

// ---------------------------------------------------------------------------
extern "C" void kernel_launch(void* const* d_in, const int* in_sizes, int n_in,
                              void* d_out, int out_size, void* d_ws,
                              size_t ws_size, hipStream_t stream) {
  (void)in_sizes; (void)n_in; (void)out_size; (void)ws_size;
  const float* x     = (const float*)d_in[0];
  const float* Wt    = (const float*)d_in[1];
  const float* Wp    = (const float*)d_in[2];
  const float* Wg    = (const float*)d_in[3];
  const float* Wz    = (const float*)d_in[4];
  const float* W_adj = (const float*)d_in[5];
  const float* g_adj = (const float*)d_in[6];
  const float* b_adj = (const float*)d_in[7];
  const float* m_adj = (const float*)d_in[8];
  const float* v_adj = (const float*)d_in[9];
  const float* W_wg  = (const float*)d_in[10];
  const float* g_wg  = (const float*)d_in[11];
  const float* b_wg  = (const float*)d_in[12];
  const float* m_wg  = (const float*)d_in[13];
  const float* v_wg  = (const float*)d_in[14];
  const float* gn_g  = (const float*)d_in[15];
  const float* gn_b  = (const float*)d_in[16];
  float* out = (float*)d_out;

  // ws layout: att|S1|S2 (3*1024 f32, zeroed per launch) | params (4096 f32)
  //            | Whi | Wlo (393216 bf16 each). Total ~1.6 MB.
  float*  att    = (float*)d_ws;
  float*  S1     = att + 1024;
  float*  S2     = S1 + 1024;
  float*  params = S2 + 1024;
  __bf16* Whi    = (__bf16*)(params + 4096);
  __bf16* Wlo    = Whi + 768 * CIN;

  k_zero<<<(3072 + 255) / 256, 256, 0, stream>>>(att, 3072);
  k_wpack<<<(768 * CIN / 2) / 256, 256, 0, stream>>>(Wt, Wp, Wg, Whi, Wlo);
  k_gemm<<<dim3(HW / NTILE, PLANES / CTILE, 4), 256, 0, stream>>>(
      x, Whi, Wlo, out, att, S1, S2);
  k_graph<<<1, 256, 0, stream>>>(att, S1, S2, Wz, W_adj, g_adj, b_adj, m_adj,
                                 v_adj, W_wg, g_wg, b_wg, m_wg, v_wg, gn_g,
                                 gn_b, params);
  k_final<<<(4u * 256u * (HW / 4)) / 256u, 256, 0, stream>>>(x, params, out);
}
